// Attention_66881230734051
// MI455X (gfx1250) — matrix operand order repaired and checked by reading, not compiled, not run on hardware
//
#include <hip/hip_runtime.h>
#include <hip/hip_bf16.h>

typedef __attribute__((ext_vector_type(16))) _Float16 v16h;
typedef __attribute__((ext_vector_type(8)))  _Float16 v8h;
typedef __attribute__((ext_vector_type(8)))  float    v8f;
typedef __attribute__((ext_vector_type(4)))  unsigned int v4u;
typedef __attribute__((ext_vector_type(8)))  int v8i;
typedef __attribute__((ext_vector_type(4)))  int v4i;

#define EMBED 1024
#define NHEAD 16
#define HD    64
#define SEQ   2048
#define BATCH 4
#define MTOT  (BATCH * SEQ)   // 8192

union HV16 { v16h v; v8h h[2]; _Float16 e[16]; };
union HV8  { v8h v; _Float16 e[8]; };
union FV8  { v8f v; float e[8]; };

static __device__ __forceinline__ v8f wmma_f16(v16h a, v16h b, v8f c) {
  // D = A(16x32 f16) * B(32x16 f16) + C(16x16 f32)
  return __builtin_amdgcn_wmma_f32_16x16x32_f16(
      /*neg_a=*/false, a, /*neg_b=*/false, b,
      /*c_mod=*/(short)0, c, /*reuse_a=*/false, /*reuse_b=*/false);
}

// ---------------------------------------------------------------------------
// Tensor Data Mover: 2D tile (f16 elements) global -> LDS.
// D# layout per CDNA5 ISA ch.8 (group0 128b, group1 256b; groups 2/3 zero).
//   tile_d0   : contiguous elements per line
//   tile_d1   : number of lines
//   stride_d0 : elements between consecutive lines in global memory
// Toolchain here uses the 6-arg builtin (extra int32x8 group + cpol).
// ---------------------------------------------------------------------------
static __device__ __forceinline__ void tdm_load_2d_f16(
    unsigned lds_off, const void* gptr,
    unsigned tile_d0, unsigned tile_d1,
    unsigned tensor_d0, unsigned tensor_d1,
    unsigned long long stride_d0) {
  unsigned long long ga = (unsigned long long)(size_t)gptr;
  v4u g0;
  g0[0] = 1u;                                    // count=1, user descriptor
  g0[1] = lds_off;                               // lds_addr (bytes)
  g0[2] = (unsigned)(ga & 0xFFFFFFFFu);          // global_addr[31:0]
  g0[3] = (unsigned)((ga >> 32) & 0x01FFFFFFu)   // global_addr[56:32]
          | 0x80000000u;                         // type = 2 ("image")
  v8i g1;
  g1[0] = (int)(1u << 16);                       // data_size=1 (2 bytes)
  g1[1] = (int)((tensor_d0 & 0xFFFFu) << 16);    // tensor_dim0[15:0]
  g1[2] = (int)(((tensor_d0 >> 16) & 0xFFFFu) |  // tensor_dim0[31:16]
                ((tensor_d1 & 0xFFFFu) << 16));  // tensor_dim1[15:0]
  g1[3] = (int)(((tensor_d1 >> 16) & 0xFFFFu) |  // tensor_dim1[31:16]
                ((tile_d0 & 0xFFFFu) << 16));    // tile_dim0
  g1[4] = (int)(tile_d1 & 0xFFFFu);              // tile_dim1 (tile_dim2=0)
  g1[5] = (int)(unsigned)(stride_d0 & 0xFFFFFFFFull);   // dim0_stride[31:0]
  g1[6] = (int)(unsigned)((stride_d0 >> 32) & 0xFFFFull);
  g1[7] = 0;
  v4i g2 = {0, 0, 0, 0};
  v4i g3 = {0, 0, 0, 0};
  v8i g4 = {0, 0, 0, 0, 0, 0, 0, 0};
  __builtin_amdgcn_tensor_load_to_lds(g0, g1, g2, g3, g4, 0);
}

// ---------------------------------------------------------------------------
// fp32 -> fp16 conversion
// ---------------------------------------------------------------------------
__global__ void f32_to_f16_kernel(const float* __restrict__ in,
                                  _Float16* __restrict__ out, int n) {
  int i = blockIdx.x * blockDim.x + threadIdx.x;
  if (i < n) out[i] = (_Float16)in[i];
}

// ---------------------------------------------------------------------------
// QKV projection: C[8192,3072] = xh[8192,1024] @ Wh[1024,3072] + b
// scattered into Q/K/V half buffers laid out [B,H,S,HD].
// Block: 256 threads (8 waves). Tile: 128(M) x 128(N). K-step 32.
// Waves: 4 over M (32 rows each) x 2 over N (64 cols each) -> 8 WMMA/K-step.
// A tile staged by TDM; B tile staged transposed by threads.
// ---------------------------------------------------------------------------
__global__ __launch_bounds__(256) void gemm_qkv_kernel(
    const _Float16* __restrict__ A,   // [8192,1024]
    const _Float16* __restrict__ B,   // [1024,3072]
    const float*    __restrict__ bias,// [3072]
    _Float16* __restrict__ Qo, _Float16* __restrict__ Ko,
    _Float16* __restrict__ Vo) {
  __shared__ _Float16 As[128][32];    // [m][k]
  __shared__ _Float16 Bst[128][32];   // transposed: [n][k]

  const int tid    = threadIdx.x;
  const int lane   = tid & 31;
  const int wid    = tid >> 5;
  const int wave_m = wid & 3;         // 4 waves over M (32 rows each)
  const int wave_n = wid >> 2;        // 2 waves over N (64 cols each)
  const int halfid = lane >> 4;       // 0/1
  const int l16    = lane & 15;
  const int m0     = blockIdx.y * 128;
  const int n0     = blockIdx.x * 128;
  const unsigned as_off = (unsigned)(size_t)&As[0][0];

  v8f acc[2][4] = {};                 // 32 x 64 per wave

  for (int k0 = 0; k0 < 1024; k0 += 32) {
    __syncthreads();
    if (wid == 0) {                   // A tile via Tensor Data Mover
      tdm_load_2d_f16(as_off, &A[(size_t)m0 * 1024 + k0],
                      /*tile*/ 32, 128, /*tensor*/ 32, 128,
                      /*stride*/ 1024ull);
    }
    {  // B tile: 32x128, stored transposed. 16 halves per thread.
      int k = tid >> 3, nb = (tid & 7) * 16;
      const _Float16* src = &B[(size_t)(k0 + k) * 3072 + n0 + nb];
      HV16 t; t.h[0] = *(const v8h*)src; t.h[1] = *(const v8h*)(src + 8);
#pragma unroll
      for (int i = 0; i < 16; ++i) Bst[nb + i][k] = t.e[i];
    }
    if (wid == 0) __builtin_amdgcn_s_wait_tensorcnt(0);
    __syncthreads();

    // A fragments (2 M-tiles): lane m = l16; K halves per ISA f16 A layout
    HV16 af[2];
#pragma unroll
    for (int mi = 0; mi < 2; ++mi) {
      const int r = wave_m * 32 + mi * 16 + l16;
      af[mi].h[0] = *(const v8h*)&As[r][halfid * 8];
      af[mi].h[1] = *(const v8h*)&As[r][16 + halfid * 8];
    }
#pragma unroll
    for (int nc = 0; nc < 4; ++nc) {
      // B fragment: lane n = l16; K = halfid*8 + j, 16 + halfid*8 + j (table order)
      const _Float16* p = &Bst[wave_n * 64 + nc * 16 + l16][halfid * 8];
      HV16 bf; bf.h[0] = *(const v8h*)p; bf.h[1] = *(const v8h*)(p + 16);
#pragma unroll
      for (int mi = 0; mi < 2; ++mi)
        acc[mi][nc] = wmma_f16(af[mi].v, bf.v, acc[mi][nc]);
    }
  }

  // Epilogue: bias add + scatter to per-head Q/K/V (half)
#pragma unroll
  for (int mi = 0; mi < 2; ++mi) {
#pragma unroll
    for (int nc = 0; nc < 4; ++nc) {
      FV8 a; a.v = acc[mi][nc];
      int col   = n0 + wave_n * 64 + nc * 16 + l16;
      int h     = col / (3 * HD);
      int rem   = col % (3 * HD);
      int which = rem / HD;
      int d     = rem % HD;
      float bv  = bias[col];
      _Float16* dst = (which == 0) ? Qo : (which == 1) ? Ko : Vo;
#pragma unroll
      for (int v = 0; v < 8; ++v) {
        int row = m0 + wave_m * 32 + mi * 16 + halfid * 8 + v;
        int bb  = row >> 11;                      // / SEQ
        int s   = row & (SEQ - 1);
        dst[((size_t)(bb * NHEAD + h) * SEQ + s) * HD + d] =
            (_Float16)(a.e[v] + bv);
      }
    }
  }
}

// ---------------------------------------------------------------------------
// Flash attention: one block = (b*h, 128-query tile); 8 waves x 16 rows.
// K chunk staged to LDS by TDM; V chunk staged transposed by threads.
// Online softmax over 32-key chunks; 8 WMMA per chunk per wave.
// ---------------------------------------------------------------------------
__global__ __launch_bounds__(256) void attn_kernel(
    const _Float16* __restrict__ Q, const _Float16* __restrict__ K,
    const _Float16* __restrict__ V, _Float16* __restrict__ O) {
  __shared__ _Float16 Ks[32][64];      // row-major key chunk: [k][d]
  __shared__ _Float16 Vt[64][32];      // transposed V chunk:  [d][k]
  __shared__ _Float16 Ps[8][16][32];   // per-wave P tile [m][k]

  const int tid    = threadIdx.x;
  const int lane   = tid & 31;
  const int wid    = tid >> 5;
  const int halfid = lane >> 4;
  const int l16    = lane & 15;
  const int bh     = blockIdx.y;                    // b*NHEAD + h
  const int q0     = blockIdx.x * 128 + wid * 16;   // this wave's query base
  const unsigned ks_off = (unsigned)(size_t)&Ks[0][0];

  const _Float16* Qbh = Q + (size_t)bh * SEQ * HD;
  const _Float16* Kbh = K + (size_t)bh * SEQ * HD;
  const _Float16* Vbh = V + (size_t)bh * SEQ * HD;

  // Q fragments (constant over the key loop): d 0..31 and 32..63
  HV16 qa0, qa1;
  {
    const _Float16* qrow = &Qbh[(size_t)(q0 + l16) * HD];
    qa0.h[0] = *(const v8h*)&qrow[halfid * 16];
    qa0.h[1] = *(const v8h*)&qrow[halfid * 16 + 8];
    qa1.h[0] = *(const v8h*)&qrow[32 + halfid * 16];
    qa1.h[1] = *(const v8h*)&qrow[40 + halfid * 16];
  }

  float rowM[8], rowL[8];
#pragma unroll
  for (int v = 0; v < 8; ++v) { rowM[v] = -1e30f; rowL[v] = 0.0f; }
  v8f o[4] = {};
  const float scale = 0.125f;  // 1/sqrt(64)

  for (int kc = 0; kc < SEQ; kc += 32) {
    __syncthreads();
    if (wid == 0) {  // K chunk (32x64 f16, contiguous rows) via TDM
      tdm_load_2d_f16(ks_off, &Kbh[(size_t)kc * HD],
                      /*tile*/ HD, 32, /*tensor*/ HD, 32, /*stride*/ HD);
    }
    {  // stage V chunk transposed into LDS
      int r = tid >> 3;            // key row 0..31
      int c = (tid & 7) * 8;       // d base
      HV8 u; u.v = *(const v8h*)&Vbh[(size_t)(kc + r) * HD + c];
#pragma unroll
      for (int i = 0; i < 8; ++i) Vt[c + i][r] = u.e[i];
    }
    if (kc + 32 < SEQ)  // global_prefetch_b8 for next key chunk
      __builtin_prefetch(&Kbh[(size_t)(kc + 32 + lane) * HD], 0, 0);
    if (wid == 0) __builtin_amdgcn_s_wait_tensorcnt(0);
    __syncthreads();

    // scores: S0 = keys kc..kc+15, S1 = keys kc+16..kc+31 (B-frags from LDS)
    v8f s0 = {}, s1 = {};
    {
      const _Float16* kr0 = &Ks[l16][halfid * 16];
      HV16 b0; b0.h[0] = *(const v8h*)kr0;        b0.h[1] = *(const v8h*)(kr0 + 8);
      HV16 b1; b1.h[0] = *(const v8h*)(kr0 + 32); b1.h[1] = *(const v8h*)(kr0 + 40);
      s0 = wmma_f16(qa0.v, b0.v, s0);
      s0 = wmma_f16(qa1.v, b1.v, s0);
      const _Float16* kr1 = &Ks[16 + l16][halfid * 16];
      HV16 b2; b2.h[0] = *(const v8h*)kr1;        b2.h[1] = *(const v8h*)(kr1 + 8);
      HV16 b3; b3.h[0] = *(const v8h*)(kr1 + 32); b3.h[1] = *(const v8h*)(kr1 + 40);
      s1 = wmma_f16(qa0.v, b2.v, s1);
      s1 = wmma_f16(qa1.v, b3.v, s1);
    }

    // online softmax; row v lives on the 16 lanes of this half-wave
    FV8 f0, f1; f0.v = s0; f1.v = s1;
    _Float16 p0h[8], p1h[8];
#pragma unroll
    for (int v = 0; v < 8; ++v) {
      float a0 = f0.e[v] * scale, a1 = f1.e[v] * scale;
      float mx = fmaxf(a0, a1);
      mx = fmaxf(mx, __shfl_xor(mx, 1));
      mx = fmaxf(mx, __shfl_xor(mx, 2));
      mx = fmaxf(mx, __shfl_xor(mx, 4));
      mx = fmaxf(mx, __shfl_xor(mx, 8));
      float newM  = fmaxf(rowM[v], mx);
      float alpha = __expf(rowM[v] - newM);
      float p0 = __expf(a0 - newM), p1 = __expf(a1 - newM);
      float rs = p0 + p1;
      rs += __shfl_xor(rs, 1);
      rs += __shfl_xor(rs, 2);
      rs += __shfl_xor(rs, 4);
      rs += __shfl_xor(rs, 8);
      rowL[v] = rowL[v] * alpha + rs;
      rowM[v] = newM;
#pragma unroll
      for (int f = 0; f < 4; ++f) o[f][v] *= alpha;
      p0h[v] = (_Float16)p0; p1h[v] = (_Float16)p1;
    }

    // C-layout -> A-layout via LDS (same-wave DS ops are in order)
#pragma unroll
    for (int v = 0; v < 8; ++v) {
      int m = halfid * 8 + v;
      Ps[wid][m][l16]      = p0h[v];
      Ps[wid][m][16 + l16] = p1h[v];
    }
    HV16 pa;
    pa.h[0] = *(const v8h*)&Ps[wid][l16][halfid * 16];
    pa.h[1] = *(const v8h*)&Ps[wid][l16][halfid * 16 + 8];

    // O += P(16x32) @ V(32x64), 4 d-chunks
#pragma unroll
    for (int f = 0; f < 4; ++f) {
      const _Float16* vp = &Vt[f * 16 + l16][halfid * 16];
      HV16 vb; vb.h[0] = *(const v8h*)vp; vb.h[1] = *(const v8h*)(vp + 8);
      o[f] = wmma_f16(pa.v, vb.v, o[f]);
    }
  }

  // normalize and write O as half, layout [B,S,E] with col = h*64+d
  const int bb = bh / NHEAD, h = bh % NHEAD;
#pragma unroll
  for (int f = 0; f < 4; ++f) {
#pragma unroll
    for (int v = 0; v < 8; ++v) {
      int m = halfid * 8 + v;
      float val = o[f][v] / rowL[v];
      O[((size_t)(bb * SEQ + q0 + m)) * EMBED + h * HD + f * 16 + l16] =
          (_Float16)val;
    }
  }
}

// ---------------------------------------------------------------------------
// Output projection: out[8192,1024] = Oh[8192,1024] @ Wouth[1024,1024] + b
// Same tiling as gemm_qkv (128x128 block, 32x64 per wave).
// ---------------------------------------------------------------------------
__global__ __launch_bounds__(256) void gemm_out_kernel(
    const _Float16* __restrict__ A,   // [8192,1024]
    const _Float16* __restrict__ B,   // [1024,1024]
    const float*    __restrict__ bias,// [1024]
    float* __restrict__ C) {          // [8192,1024] fp32
  __shared__ _Float16 As[128][32];
  __shared__ _Float16 Bst[128][32];

  const int tid    = threadIdx.x;
  const int lane   = tid & 31;
  const int wid    = tid >> 5;
  const int wave_m = wid & 3;
  const int wave_n = wid >> 2;
  const int halfid = lane >> 4;
  const int l16    = lane & 15;
  const int m0     = blockIdx.y * 128;
  const int n0     = blockIdx.x * 128;
  const unsigned as_off = (unsigned)(size_t)&As[0][0];

  v8f acc[2][4] = {};

  for (int k0 = 0; k0 < 1024; k0 += 32) {
    __syncthreads();
    if (wid == 0) {
      tdm_load_2d_f16(as_off, &A[(size_t)m0 * 1024 + k0],
                      /*tile*/ 32, 128, /*tensor*/ 32, 128,
                      /*stride*/ 1024ull);
    }
    {
      int k = tid >> 3, nb = (tid & 7) * 16;
      const _Float16* src = &B[(size_t)(k0 + k) * 1024 + n0 + nb];
      HV16 t; t.h[0] = *(const v8h*)src; t.h[1] = *(const v8h*)(src + 8);
#pragma unroll
      for (int i = 0; i < 16; ++i) Bst[nb + i][k] = t.e[i];
    }
    if (wid == 0) __builtin_amdgcn_s_wait_tensorcnt(0);
    __syncthreads();

    HV16 af[2];
#pragma unroll
    for (int mi = 0; mi < 2; ++mi) {
      const int r = wave_m * 32 + mi * 16 + l16;
      af[mi].h[0] = *(const v8h*)&As[r][halfid * 8];
      af[mi].h[1] = *(const v8h*)&As[r][16 + halfid * 8];
    }
#pragma unroll
    for (int nc = 0; nc < 4; ++nc) {
      const _Float16* p = &Bst[wave_n * 64 + nc * 16 + l16][halfid * 8];
      HV16 bf; bf.h[0] = *(const v8h*)p; bf.h[1] = *(const v8h*)(p + 16);
#pragma unroll
      for (int mi = 0; mi < 2; ++mi)
        acc[mi][nc] = wmma_f16(af[mi].v, bf.v, acc[mi][nc]);
    }
  }

#pragma unroll
  for (int mi = 0; mi < 2; ++mi) {
#pragma unroll
    for (int nc = 0; nc < 4; ++nc) {
      FV8 a; a.v = acc[mi][nc];
      int col  = n0 + wave_n * 64 + nc * 16 + l16;
      float bv = bias[col];
#pragma unroll
      for (int v = 0; v < 8; ++v) {
        int row = m0 + wave_m * 32 + mi * 16 + halfid * 8 + v;
        C[(size_t)row * 1024 + col] = a.e[v] + bv;
      }
    }
  }
}

// ---------------------------------------------------------------------------
extern "C" void kernel_launch(void* const* d_in, const int* in_sizes, int n_in,
                              void* d_out, int out_size, void* d_ws,
                              size_t ws_size, hipStream_t stream) {
  const float* x     = (const float*)d_in[0];  // [4,2048,1024]
  const float* Wqkv  = (const float*)d_in[1];  // [1024,3072]
  const float* bqkv  = (const float*)d_in[2];  // [3072]
  const float* Wout  = (const float*)d_in[3];  // [1024,1024]
  const float* bout  = (const float*)d_in[4];  // [1024]
  float* out = (float*)d_out;

  char* ws = (char*)d_ws;
  size_t off = 0;
  auto alloc = [&](size_t bytes) -> void* {
    void* p = ws + off;
    off += (bytes + 255) & ~(size_t)255;
    return p;
  };
  const size_t nX  = (size_t)MTOT * EMBED;        // 8388608
  const size_t nW1 = (size_t)EMBED * 3 * EMBED;   // 3145728
  const size_t nW2 = (size_t)EMBED * EMBED;       // 1048576
  const size_t nQ  = (size_t)BATCH * NHEAD * SEQ * HD;  // 8388608

  _Float16* xh    = (_Float16*)alloc(nX * 2);
  _Float16* Wqkvh = (_Float16*)alloc(nW1 * 2);
  _Float16* Wouth = (_Float16*)alloc(nW2 * 2);
  _Float16* Qh    = (_Float16*)alloc(nQ * 2);
  _Float16* Kh    = (_Float16*)alloc(nQ * 2);
  _Float16* Vh    = (_Float16*)alloc(nQ * 2);
  _Float16* Oh    = (_Float16*)alloc(nX * 2);

  f32_to_f16_kernel<<<(int)(nX  / 256), 256, 0, stream>>>(x,    xh,    (int)nX);
  f32_to_f16_kernel<<<(int)(nW1 / 256), 256, 0, stream>>>(Wqkv, Wqkvh, (int)nW1);
  f32_to_f16_kernel<<<(int)(nW2 / 256), 256, 0, stream>>>(Wout, Wouth, (int)nW2);

  gemm_qkv_kernel<<<dim3(3 * EMBED / 128, MTOT / 128), 256, 0, stream>>>(
      xh, Wqkvh, bqkv, Qh, Kh, Vh);

  attn_kernel<<<dim3(SEQ / 128, BATCH * NHEAD), 256, 0, stream>>>(
      Qh, Kh, Vh, Oh);

  gemm_out_kernel<<<dim3(EMBED / 128, MTOT / 128), 256, 0, stream>>>(
      Oh, Wouth, bout, out);
}
